// EdgeWeight_42691974922541
// MI455X (gfx1250) — compile-verified, matrix operand order checked by reading, not run on hardware
//
#include <hip/hip_runtime.h>
#include <hip/hip_bf16.h>

typedef __attribute__((ext_vector_type(16))) __bf16 v16bf;
typedef __attribute__((ext_vector_type(8)))  float  v8f;
typedef __attribute__((ext_vector_type(4)))  float  v4f;

#define BB 64
#define NN 128
#define DD 128
#define HH 64
#define KK 16

// ---------------------------------------------------------------------------
// Prep: weight-norm w1 -> bf16 in WMMA B-matrix lane layout; weight-norm w2.
// ws layout: [0 .. 16KB) = w1 bf16 tiles, [16KB ..) = w2 f32 (64).
// B-tile layout for 16-bit B (KxN = 32x16): lane L holds column N = L&15;
// lanes>=16 shift K by 8. Element e<8 -> K = kc*32 + kb + e,
// e>=8 -> K = kc*32 + 16 + kb + (e-8).  Linear: ((n*4+kc)*32 + lane)*16 + e.
// ---------------------------------------------------------------------------
__global__ void EdgeWeight_prep_kernel(const float* __restrict__ v1,
                                       const float* __restrict__ g1,
                                       const float* __restrict__ v2,
                                       const float* __restrict__ g2,
                                       __bf16* __restrict__ wsW1,
                                       float*  __restrict__ wsW2) {
  const int h = threadIdx.x;  // 64 threads, one per hidden unit
  if (h >= HH) return;

  // row norm of v1[h, :]
  float ss = 0.f;
  for (int d = 0; d < DD; ++d) {
    float x = v1[h * DD + d];
    ss += x * x;
  }
  const float scale = g1[h] * __frsqrt_rn(ss);

  // w2 (every thread redundantly computes the 64-element norm; trivial)
  float ss2 = 0.f;
  for (int l = 0; l < HH; ++l) {
    float x = v2[l];
    ss2 += x * x;
  }
  wsW2[h] = v2[h] * g2[0] * __frsqrt_rn(ss2);

  const int n   = h >> 4;
  const int col = h & 15;
  for (int kc = 0; kc < 4; ++kc) {
    for (int half = 0; half < 2; ++half) {
      const int lane = col + 16 * half;
      const int kb   = 8 * half;
      const int base = ((n * 4 + kc) * 32 + lane) * 16;
      for (int e = 0; e < 8; ++e) {
        wsW1[base + e]     = (__bf16)(v1[h * DD + kc * 32 + kb + e] * scale);
        wsW1[base + 8 + e] = (__bf16)(v1[h * DD + kc * 32 + 16 + kb + e] * scale);
      }
    }
  }
}

// ---------------------------------------------------------------------------
// Main: one block per (b,i). 256 threads = 8 waves; wave w owns j-rows
// [16w, 16w+16). GEMM h = relu(feats @ w1^T + b1) via bf16 WMMA (B tiles
// staged in LDS -> DScnt, decoupled from the NT feats stream on LOADcnt),
// score = h.w2, softmax over j, exact top-16 threshold, masked write.
// ---------------------------------------------------------------------------
__global__ __launch_bounds__(256) void
EdgeWeight_main_kernel(const float* __restrict__ feats,
                       const float* __restrict__ b1,
                       const __bf16* __restrict__ wsW1,
                       const float*  __restrict__ wsW2,
                       float* __restrict__ out) {
  const int tid     = threadIdx.x;
  const int lane    = tid & 31;
  const int wave    = tid >> 5;
  const int col     = lane & 15;
  const int halfSel = lane >> 4;     // 0 or 1
  const int kb      = halfSel * 8;   // K shift for upper half-lanes
  const int jBase   = wave * 16;

  __shared__ __align__(16) __bf16 sW1[HH * DD];  // 16 KB of B tiles
  __shared__ float sScores[NN];
  __shared__ float sWeights[NN];
  __shared__ float sThresh;

  // ---- stage w1 tiles into LDS (8192 bf16 = 1024 uint4; 4 per thread)
  {
    const uint4* src = (const uint4*)wsW1;
    uint4*       dst = (uint4*)sW1;
#pragma unroll
    for (int c = 0; c < 4; ++c) dst[c * 256 + tid] = src[c * 256 + tid];
  }
  __syncthreads();

  // ---- GEMM: acc[n] = 16x16 f32 tile (j x h), n-tile n covers h = 16n..16n+15
  v8f zero = {};
  v8f acc[4] = {zero, zero, zero, zero};

  float b1v[4], w2v[4];
#pragma unroll
  for (int n = 0; n < 4; ++n) {
    b1v[n] = b1[n * 16 + col];
    w2v[n] = wsW2[n * 16 + col];
  }

  // A row for this lane: j = jBase + col  (16-bit A layout: lane = M)
  const size_t rowBase = ((size_t)blockIdx.x * NN + (size_t)(jBase + col)) * DD;
  const float* rowPtr  = feats + rowBase;

#pragma unroll
  for (int kc = 0; kc < 4; ++kc) {
    // load 16 f32 for this lane's A fragment (non-temporal: one-pass stream),
    // convert to bf16
    const float* p = rowPtr + kc * 32 + kb;
    v4f f0 = __builtin_nontemporal_load((const v4f*)(p));
    v4f f1 = __builtin_nontemporal_load((const v4f*)(p + 4));
    v4f f2 = __builtin_nontemporal_load((const v4f*)(p + 16));
    v4f f3 = __builtin_nontemporal_load((const v4f*)(p + 20));
    v16bf a;
    a[0]  = (__bf16)f0[0];  a[1]  = (__bf16)f0[1];
    a[2]  = (__bf16)f0[2];  a[3]  = (__bf16)f0[3];
    a[4]  = (__bf16)f1[0];  a[5]  = (__bf16)f1[1];
    a[6]  = (__bf16)f1[2];  a[7]  = (__bf16)f1[3];
    a[8]  = (__bf16)f2[0];  a[9]  = (__bf16)f2[1];
    a[10] = (__bf16)f2[2];  a[11] = (__bf16)f2[3];
    a[12] = (__bf16)f3[0];  a[13] = (__bf16)f3[1];
    a[14] = (__bf16)f3[2];  a[15] = (__bf16)f3[3];

#pragma unroll
    for (int n = 0; n < 4; ++n) {
      v16bf bt = *(const v16bf*)&sW1[((n * 4 + kc) * 32 + lane) * 16];
      acc[n] = __builtin_amdgcn_wmma_f32_16x16x32_bf16(
          false, a, false, bt, (short)0, acc[n], false, false);
    }
  }

  // ---- per-row epilogue: relu(acc + b1) . w2, reduce across 16 lanes
  // C layout: VGPR r, lanes 0-15 -> row M=r; lanes 16-31 -> row M=r+8; N=lane&15
#pragma unroll
  for (int r = 0; r < 8; ++r) {
    float partial = 0.f;
#pragma unroll
    for (int n = 0; n < 4; ++n) {
      float hv = acc[n][r] + b1v[n];
      hv = fmaxf(hv, 0.f);
      partial += hv * w2v[n];
    }
    partial += __shfl_xor(partial, 1, 32);
    partial += __shfl_xor(partial, 2, 32);
    partial += __shfl_xor(partial, 4, 32);
    partial += __shfl_xor(partial, 8, 32);
    if (col == 0) sScores[jBase + halfSel * 8 + r] = partial;
  }
  __syncthreads();

  // ---- softmax over 128 scores (wave 0; b2 shift is softmax-invariant)
  if (wave == 0) {
    float s0 = sScores[lane];
    float s1 = sScores[lane + 32];
    float s2 = sScores[lane + 64];
    float s3 = sScores[lane + 96];
    float mx = fmaxf(fmaxf(s0, s1), fmaxf(s2, s3));
#pragma unroll
    for (int off = 1; off < 32; off <<= 1)
      mx = fmaxf(mx, __shfl_xor(mx, off, 32));
    float e0 = __expf(s0 - mx), e1 = __expf(s1 - mx);
    float e2 = __expf(s2 - mx), e3 = __expf(s3 - mx);
    float sum = e0 + e1 + e2 + e3;
#pragma unroll
    for (int off = 1; off < 32; off <<= 1)
      sum += __shfl_xor(sum, off, 32);
    float inv = 1.f / sum;
    sWeights[lane]      = e0 * inv;
    sWeights[lane + 32] = e1 * inv;
    sWeights[lane + 64] = e2 * inv;
    sWeights[lane + 96] = e3 * inv;
  }
  __syncthreads();

  // ---- exact top-16 threshold: value at descending-sorted index 15.
  // For value v: cg = #strictly-greater, ce = #equal. v sits at sorted
  // positions [cg, cg+ce); it is index 15 iff cg <= 15 < cg+ce.
  float wj = 0.f;
  if (tid < NN) {
    wj = sWeights[tid];
    int cg = 0, ce = 0;
    for (int l = 0; l < NN; ++l) {
      float v = sWeights[l];
      cg += (v > wj) ? 1 : 0;
      ce += (v == wj) ? 1 : 0;
    }
    if (cg <= (KK - 1) && cg + ce > (KK - 1)) sThresh = wj;  // all writers agree
  }
  __syncthreads();

  if (tid < NN) {
    const int irow = blockIdx.x & (NN - 1);
    float o = (wj >= sThresh || tid == irow) ? wj : 0.f;
    out[(size_t)blockIdx.x * NN + tid] = o;
  }
}

// ---------------------------------------------------------------------------
extern "C" void kernel_launch(void* const* d_in, const int* in_sizes, int n_in,
                              void* d_out, int out_size, void* d_ws, size_t ws_size,
                              hipStream_t stream) {
  const float* feats = (const float*)d_in[0];
  const float* v1    = (const float*)d_in[1];
  const float* g1    = (const float*)d_in[2];
  const float* b1    = (const float*)d_in[3];
  const float* v2    = (const float*)d_in[4];
  const float* g2    = (const float*)d_in[5];
  // d_in[6] = b2 (softmax-invariant shift, unused), d_in[7] = k (hardcoded 16)
  float* out = (float*)d_out;

  __bf16* wsW1 = (__bf16*)d_ws;                       // 64*128 bf16 = 16 KB
  float*  wsW2 = (float*)((char*)d_ws + HH * DD * 2); // 64 f32

  EdgeWeight_prep_kernel<<<1, 64, 0, stream>>>(v1, g1, v2, g2, wsW1, wsW2);
  EdgeWeight_main_kernel<<<BB * NN, 256, 0, stream>>>(feats, b1, wsW1, wsW2, out);
}
